// SelfAttentionHead_31104153158187
// MI455X (gfx1250) — matrix operand order repaired and checked by reading, not compiled, not run on hardware
//
#include <hip/hip_runtime.h>
#include <hip/hip_bf16.h>
#include <math.h>

#define B_  8
#define T_  2048
#define E_  1024
#define H_  128

typedef __attribute__((ext_vector_type(16))) __bf16   bf16x16;
typedef __attribute__((ext_vector_type(8)))  float    f32x8;
typedef __attribute__((ext_vector_type(4)))  unsigned uint4v;
typedef __attribute__((ext_vector_type(4)))  float    float4v;

union BFrag {
    bf16x16        v;
    unsigned short us[16];
    uint4v         q[2];
};

__device__ __forceinline__ unsigned short f2bf(float f) {
    unsigned u = __float_as_uint(f);
    unsigned r = u + 0x7FFFu + ((u >> 16) & 1u);
    return (unsigned short)(r >> 16);
}

__device__ __forceinline__ f32x8 wmma_bf16(const BFrag& a, const BFrag& b, f32x8 c) {
    return __builtin_amdgcn_wmma_f32_16x16x32_bf16(
        /*neg_a=*/false, a.v, /*neg_b=*/false, b.v,
        /*c_mod=*/(short)0, c, /*reuse_a=*/false, /*reuse_b=*/false);
}

// ---------------------------------------------------------------------------
// K0: weights [E][H] fp32 -> Wt [3][H][E] bf16 (transposed, B-frag friendly)
// ---------------------------------------------------------------------------
__global__ void __launch_bounds__(256) k_wt(
    const float* __restrict__ Wq, const float* __restrict__ Wk,
    const float* __restrict__ Wv, unsigned short* __restrict__ Wt)
{
    int idx = blockIdx.x * 256 + threadIdx.x;
    if (idx >= 3 * H_ * E_) return;
    int mat = idx / (H_ * E_);
    int rem = idx % (H_ * E_);
    int n = rem / E_;
    int k = rem % E_;
    const float* W = (mat == 0) ? Wq : ((mat == 1) ? Wk : Wv);
    Wt[idx] = f2bf(W[(size_t)k * H_ + n]);
}

// ---------------------------------------------------------------------------
// K1: projections. grid = (rowblocks of 128, matrix 0=q 1=k 2=v), 256 thr.
// wave w handles 16 rows x 128 cols, K-loop over E in steps of 32.
// ---------------------------------------------------------------------------
__global__ void __launch_bounds__(256) k_proj(
    const float* __restrict__ x,
    const unsigned short* __restrict__ Wt,   // [3][H][E] bf16
    unsigned short* __restrict__ qb,         // [B*T][H] bf16
    unsigned short* __restrict__ kb,         // [B*T][H] bf16
    float* __restrict__ vf)                  // [B*T][H] f32
{
    const int tid  = threadIdx.x;
    const int lane = tid & 31, w = tid >> 5;
    const int mat  = blockIdx.y;
    const int r0   = blockIdx.x * 128 + w * 16;
    const int mm   = lane & 15, kh = lane >> 4;

    f32x8 acc[8] = {};

    const unsigned short* wbase = Wt + (size_t)mat * H_ * E_;
    const float* xr = x + (size_t)(r0 + mm) * E_;

    for (int k0 = 0; k0 < E_; k0 += 32) {
        BFrag a;
        {
            const float* p0 = xr + k0 + kh * 8;
            float4v f0 = *(const float4v*)(p0);
            float4v f1 = *(const float4v*)(p0 + 4);
            float4v f2 = *(const float4v*)(p0 + 16);
            float4v f3 = *(const float4v*)(p0 + 20);
            a.us[0]  = f2bf(f0.x); a.us[1]  = f2bf(f0.y);
            a.us[2]  = f2bf(f0.z); a.us[3]  = f2bf(f0.w);
            a.us[4]  = f2bf(f1.x); a.us[5]  = f2bf(f1.y);
            a.us[6]  = f2bf(f1.z); a.us[7]  = f2bf(f1.w);
            a.us[8]  = f2bf(f2.x); a.us[9]  = f2bf(f2.y);
            a.us[10] = f2bf(f2.z); a.us[11] = f2bf(f2.w);
            a.us[12] = f2bf(f3.x); a.us[13] = f2bf(f3.y);
            a.us[14] = f2bf(f3.z); a.us[15] = f2bf(f3.w);
        }
        #pragma unroll
        for (int nt = 0; nt < 8; ++nt) {
            BFrag b;
            const unsigned short* wp =
                wbase + (size_t)(nt * 16 + mm) * E_ + k0 + kh * 8;
            b.q[0] = *(const uint4v*)wp;
            b.q[1] = *(const uint4v*)(wp + 16);
            acc[nt] = wmma_bf16(a, b, acc[nt]);
        }
    }

    // writeback: D layout -> lane col = mm, rows kh*8+r
    #pragma unroll
    for (int nt = 0; nt < 8; ++nt) {
        #pragma unroll
        for (int r = 0; r < 8; ++r) {
            int rg = r0 + kh * 8 + r;
            size_t o = (size_t)rg * H_ + nt * 16 + mm;
            float vv = acc[nt][r];
            if (mat == 0)      qb[o] = f2bf(vv);
            else if (mat == 1) kb[o] = f2bf(vv);
            else               vf[o] = vv;
        }
    }
}

// ---------------------------------------------------------------------------
// K2: per-column softmax stats over causal QK^T; grid=(T/32 col blocks, B),
// 256 thr = 8 waves striding row tiles. K fragments for the fixed 32-column
// block are hoisted into registers (loop-invariant). Then
// vs[b][h][j] = v[j][h]/denom_j (bf16, transposed for PV B-fragments).
// ---------------------------------------------------------------------------
__global__ void __launch_bounds__(256) k_stats(
    const unsigned short* __restrict__ qb,
    const unsigned short* __restrict__ kb,
    const float* __restrict__ vf,
    unsigned short* __restrict__ vst,   // [B][H][T] bf16
    float* __restrict__ mb)             // [B][T]
{
    __shared__ float redm[8][32];
    __shared__ float redd[8][32];
    __shared__ float sinv[32];

    const int tid = threadIdx.x, lane = tid & 31, w = tid >> 5;
    const int b = blockIdx.y, jb = blockIdx.x;
    const int jstart = jb * 32;
    const int nn = lane & 15, kh = lane >> 4;

    const unsigned short* qbase = qb + (size_t)b * T_ * H_;
    const unsigned short* kbase = kb + (size_t)b * T_ * H_;

    // Hoist the K-block fragments: 2 column tiles x 4 K-steps (64 VGPRs).
    BFrag bk0[4], bk1[4];
    #pragma unroll
    for (int ks = 0; ks < 4; ++ks) {
        const int k0 = ks * 32;
        const unsigned short* pb0 =
            kbase + (size_t)(jstart + nn) * H_ + k0 + kh * 16;
        bk0[ks].q[0] = *(const uint4v*)pb0;
        bk0[ks].q[1] = *(const uint4v*)(pb0 + 8);
        const unsigned short* pb1 =
            kbase + (size_t)(jstart + 16 + nn) * H_ + k0 + kh * 16;
        bk1[ks].q[0] = *(const uint4v*)pb1;
        bk1[ks].q[1] = *(const uint4v*)(pb1 + 8);
    }

    float mrun0 = -INFINITY, mrun1 = -INFINITY;
    float drun0 = 0.f, drun1 = 0.f;

    const int ntiles = (T_ - jstart) >> 4;
    for (int it = w; it < ntiles; it += 8) {
        const int i0 = jstart + it * 16;
        f32x8 s0 = {}, s1 = {};
        #pragma unroll
        for (int ks = 0; ks < 4; ++ks) {
            BFrag a;
            const unsigned short* pa =
                qbase + (size_t)(i0 + nn) * H_ + ks * 32 + kh * 16;
            a.q[0] = *(const uint4v*)pa;
            a.q[1] = *(const uint4v*)(pa + 8);
            s0 = wmma_bf16(a, bk0[ks], s0);
            s1 = wmma_bf16(a, bk1[ks], s1);
        }
        // online column-softmax update; D layout: lane col nn, rows kh*8+r
        #pragma unroll
        for (int c = 0; c < 2; ++c) {
            const f32x8 sv = c ? s1 : s0;
            const int jg = jstart + c * 16 + nn;
            float mt = -INFINITY;
            #pragma unroll
            for (int r = 0; r < 8; ++r) {
                int ig = i0 + kh * 8 + r;
                mt = (ig >= jg) ? fmaxf(mt, sv[r]) : mt;
            }
            mt = fmaxf(mt, __shfl_xor(mt, 16, 32));
            float st = 0.f;
            #pragma unroll
            for (int r = 0; r < 8; ++r) {
                int ig = i0 + kh * 8 + r;
                st += (ig >= jg) ? __expf(sv[r] - mt) : 0.f;
            }
            st += __shfl_xor(st, 16, 32);
            float& mrun = c ? mrun1 : mrun0;
            float& drun = c ? drun1 : drun0;
            float mnew = fmaxf(mrun, mt);
            float ea = (mrun > -INFINITY) ? __expf(mrun - mnew) : 0.f;
            float eb = (mt   > -INFINITY) ? __expf(mt   - mnew) : 0.f;
            drun = drun * ea + st * eb;
            mrun = mnew;
        }
    }
    if (lane < 16) {
        redm[w][lane]      = mrun0; redd[w][lane]      = drun0;
        redm[w][16 + lane] = mrun1; redd[w][16 + lane] = drun1;
    }
    __syncthreads();
    if (tid < 32) {
        float mmax = -INFINITY;
        for (int ww = 0; ww < 8; ++ww) mmax = fmaxf(mmax, redm[ww][tid]);
        float dd = 0.f;
        for (int ww = 0; ww < 8; ++ww)
            dd += (redm[ww][tid] > -INFINITY)
                      ? redd[ww][tid] * __expf(redm[ww][tid] - mmax) : 0.f;
        mb[(size_t)b * T_ + jstart + tid] = mmax;
        sinv[tid] = 1.0f / dd;      // diagonal term guarantees dd >= 1
    }
    __syncthreads();
    {
        const int jj = tid >> 3;
        const int hb = (tid & 7) * 16;
        const int jgl = jstart + jj;
        const float is = sinv[jj];
        const float* vr = vf + ((size_t)b * T_ + jgl) * H_ + hb;
        unsigned short* vo = vst + (size_t)b * H_ * T_ + jgl;
        #pragma unroll
        for (int h = 0; h < 16; ++h)
            vo[(size_t)(hb + h) * T_] = f2bf(vr[h] * is);
    }
}

// ---------------------------------------------------------------------------
// K3: out = exp(QK^T - m_j) @ Vs, causal. grid=(T/64 row blocks, B),
// 128 thr = 4 waves, one 16-row tile each. Q fragments are hoisted into
// registers (invariant over the j-loop); P bounced via per-wave LDS tile
// (same-wave LDS is in-order, so no barrier -> triangular trip counts OK).
// ---------------------------------------------------------------------------
__global__ void __launch_bounds__(128) k_attn(
    const unsigned short* __restrict__ qb,
    const unsigned short* __restrict__ kb,
    const unsigned short* __restrict__ vst,  // [B][H][T] bf16 (pre-scaled)
    const float* __restrict__ mb,            // [B][T]
    float* __restrict__ out)                 // [B][T][H] f32
{
    __shared__ unsigned short Plds[4][16 * 32];
    const int tid = threadIdx.x, lane = tid & 31, w = tid >> 5;
    const int b = blockIdx.y, rb = blockIdx.x;
    const int i0 = rb * 64 + w * 16;
    const int nn = lane & 15, kh = lane >> 4;

    const unsigned short* qbase = qb  + (size_t)b * T_ * H_;
    const unsigned short* kbase = kb  + (size_t)b * T_ * H_;
    const unsigned short* vbase = vst + (size_t)b * H_ * T_;
    const float* mbase = mb + (size_t)b * T_;
    unsigned short* myP = &Plds[w][0];

    // Hoist Q fragments for this wave's 16 rows: 4 K-steps (32 VGPRs).
    BFrag aq[4];
    #pragma unroll
    for (int ks = 0; ks < 4; ++ks) {
        const unsigned short* pa =
            qbase + (size_t)(i0 + nn) * H_ + ks * 32 + kh * 16;
        aq[ks].q[0] = *(const uint4v*)pa;
        aq[ks].q[1] = *(const uint4v*)(pa + 8);
    }

    f32x8 o[8] = {};

    for (int j0 = 0; j0 < i0 + 16; j0 += 32) {
        f32x8 s0 = {}, s1 = {};
        #pragma unroll
        for (int ks = 0; ks < 4; ++ks) {
            const int k0 = ks * 32;
            BFrag b0, b1;
            const unsigned short* pb0 =
                kbase + (size_t)(j0 + nn) * H_ + k0 + kh * 16;
            b0.q[0] = *(const uint4v*)pb0;
            b0.q[1] = *(const uint4v*)(pb0 + 8);
            const unsigned short* pb1 =
                kbase + (size_t)(j0 + 16 + nn) * H_ + k0 + kh * 16;
            b1.q[0] = *(const uint4v*)pb1;
            b1.q[1] = *(const uint4v*)(pb1 + 8);
            s0 = wmma_bf16(aq[ks], b0, s0);
            s1 = wmma_bf16(aq[ks], b1, s1);
        }
        // P = exp(S - m_col) with causal mask, D-layout -> LDS [i][j] tile
        #pragma unroll
        for (int c = 0; c < 2; ++c) {
            const f32x8 sv = c ? s1 : s0;
            const int jg = j0 + c * 16 + nn;
            const float mcol = mbase[jg];
            #pragma unroll
            for (int r = 0; r < 8; ++r) {
                const int ig = i0 + kh * 8 + r;
                float p = (ig >= jg) ? __expf(sv[r] - mcol) : 0.f;
                myP[(kh * 8 + r) * 32 + c * 16 + nn] = f2bf(p);
            }
        }
        asm volatile("" ::: "memory");   // keep LDS store->load order in IR
        BFrag pf;
        {
            const unsigned short* pp = myP + nn * 32 + kh * 16;
            pf.q[0] = *(const uint4v*)pp;
            pf.q[1] = *(const uint4v*)(pp + 8);
        }
        #pragma unroll
        for (int ht = 0; ht < 8; ++ht) {
            BFrag bv;
            const unsigned short* pv =
                vbase + (size_t)(ht * 16 + nn) * T_ + j0 + kh * 16;
            bv.q[0] = *(const uint4v*)pv;
            bv.q[1] = *(const uint4v*)(pv + 8);
            o[ht] = wmma_bf16(pf, bv, o[ht]);
        }
        asm volatile("" ::: "memory");
    }

    float* obase = out + ((size_t)b * T_ + i0) * H_;
    #pragma unroll
    for (int ht = 0; ht < 8; ++ht) {
        #pragma unroll
        for (int r = 0; r < 8; ++r)
            obase[(size_t)(kh * 8 + r) * H_ + ht * 16 + nn] = o[ht][r];
    }
}

// ---------------------------------------------------------------------------
extern "C" void kernel_launch(void* const* d_in, const int* in_sizes, int n_in,
                              void* d_out, int out_size, void* d_ws, size_t ws_size,
                              hipStream_t stream) {
    const float* x  = (const float*)d_in[0];
    const float* Wk = (const float*)d_in[1];
    const float* Wq = (const float*)d_in[2];
    const float* Wv = (const float*)d_in[3];
    float* out = (float*)d_out;

    char* ws = (char*)d_ws;
    size_t o_qb  = 0;
    size_t o_kb  = o_qb  + (size_t)B_ * T_ * H_ * 2;   // q bf16
    size_t o_v   = o_kb  + (size_t)B_ * T_ * H_ * 2;   // k bf16
    size_t o_vst = o_v   + (size_t)B_ * T_ * H_ * 4;   // v f32
    size_t o_m   = o_vst + (size_t)B_ * H_ * T_ * 2;   // vs bf16 (transposed)
    size_t o_wt  = o_m   + (size_t)B_ * T_ * 4;        // column max
    // total ~21.8 MB of workspace

    unsigned short* qbp = (unsigned short*)(ws + o_qb);
    unsigned short* kbp = (unsigned short*)(ws + o_kb);
    float*          vfp = (float*)(ws + o_v);
    unsigned short* vsp = (unsigned short*)(ws + o_vst);
    float*          mbp = (float*)(ws + o_m);
    unsigned short* wtp = (unsigned short*)(ws + o_wt);

    k_wt   <<<dim3((3 * H_ * E_ + 255) / 256), 256, 0, stream>>>(Wq, Wk, Wv, wtp);
    k_proj <<<dim3((B_ * T_) / 128, 3),        256, 0, stream>>>(x, wtp, qbp, kbp, vfp);
    k_stats<<<dim3(T_ / 32, B_),               256, 0, stream>>>(qbp, kbp, vfp, vsp, mbp);
    k_attn <<<dim3(T_ / 64, B_),               128, 0, stream>>>(qbp, kbp, vsp, mbp, out);
}
